// GroupedQueryAttention_36756330119712
// MI455X (gfx1250) — matrix operand order repaired and checked by reading, not compiled, not run on hardware
//
#include <hip/hip_runtime.h>
#include <hip/hip_bf16.h>
#include <math.h>
#include <stdint.h>

// ---------------------------------------------------------------------------
// GQA forward for MI455X (gfx1250): bf16 WMMA pipeline with async-to-LDS
// double-buffered GEMMs (explicitly two-phase so LDS reads stay in AS(3)).
//   B=2, S=2048, D_MODEL=2048, H=32, KV=8, DK=64
// ---------------------------------------------------------------------------

typedef __attribute__((ext_vector_type(16))) __bf16 v16bf;
typedef __attribute__((ext_vector_type(8)))  __bf16 v8bf;
typedef __attribute__((ext_vector_type(8)))  float  v8f;

#define WMMA_BF16(A, B, C) \
  __builtin_amdgcn_wmma_f32_16x16x32_bf16(false, (A), false, (B), (short)0, (C), false, false)

// CDNA5 async copy global->LDS (ASYNCcnt-tracked). VDST = per-lane LDS byte
// address (low 32 bits of generic pointer), VADDR = 64-bit global address.
#define ASYNC_B128(ldsaddr, gaddr)                                        \
  asm volatile("global_load_async_to_lds_b128 %0, %1, off"                \
               :: "v"(ldsaddr), "v"(gaddr) : "memory")

#define WAIT_ASYNC_4() asm volatile("s_wait_asynccnt 4" ::: "memory")
#define WAIT_ASYNC_0() asm volatile("s_wait_asynccnt 0" ::: "memory")

static __device__ __forceinline__ v16bf frag_cat(v8bf lo, v8bf hi) {
  return __builtin_shufflevector(lo, hi, 0,1,2,3,4,5,6,7,8,9,10,11,12,13,14,15);
}

// A-fragment, 16(M) x 32(K) bf16, row-major rows = M, `stride` elems.
// lane L -> M = L&15; VGPRs 0-3: K = 8*half + {0..7}; VGPRs 4-7: +16.
static __device__ __forceinline__ v16bf load_frag_a(const __bf16* base, int stride,
                                                    int kk, int lane) {
  int r = lane & 15, h = lane >> 4;
  const __bf16* p = base + (size_t)r * stride + kk + h * 8;
  v8bf lo = *(const v8bf*)p;
  v8bf hi = *(const v8bf*)(p + 16);
  return frag_cat(lo, hi);
}

// B-fragment, 32(K) x 16(N) bf16, from N-major storage (rows = N columns).
// lane L -> N = L&15; VGPRs 0-3: K = 8*half + {0..7}; VGPRs 4-7: +16.
static __device__ __forceinline__ v16bf load_frag_b(const __bf16* baseT, int stride,
                                                    int kk, int lane) {
  int n = lane & 15, h = lane >> 4;
  const __bf16* p = baseT + (size_t)n * stride + kk + h * 8;
  v8bf lo = *(const v8bf*)p;
  v8bf hi = *(const v8bf*)(p + 16);
  return frag_cat(lo, hi);
}

// ---------------------------------------------------------------------------
// Async staging of one K-slice (depth 32) of A (128 rows) and B^T (128 rows)
// into LDS. 256 threads; each thread issues 2+2 b128 async copies (16B each).
// LDS layout: row-major [128][32] bf16 (64B rows).
// ---------------------------------------------------------------------------
static __device__ __forceinline__ void stage_tiles(
    const __bf16* Ablk, int lda, const __bf16* Bblk, int ldb,
    __bf16* Asmb, __bf16* Bsmb, int kk, int tid) {
#pragma unroll
  for (int i = 0; i < 2; ++i) {
    int c = tid * 2 + i;            // 0..511 16B-chunks
    int row = c >> 2, seg = c & 3;  // 128 rows x 4 chunks
    unsigned la = (unsigned)(uintptr_t)(Asmb + c * 8);
    unsigned long long ga =
        (unsigned long long)(uintptr_t)(Ablk + (size_t)row * lda + kk + seg * 8);
    ASYNC_B128(la, ga);
  }
#pragma unroll
  for (int i = 0; i < 2; ++i) {
    int c = tid * 2 + i;
    int row = c >> 2, seg = c & 3;
    unsigned la = (unsigned)(uintptr_t)(Bsmb + c * 8);
    unsigned long long ga =
        (unsigned long long)(uintptr_t)(Bblk + (size_t)row * ldb + kk + seg * 8);
    ASYNC_B128(la, ga);
  }
}

// Consume one staged K-slice: 8 WMMAs per wave (wave tile 32M x 64N).
static __device__ __forceinline__ void compute_slice(
    const __bf16* Asb, const __bf16* Bsb, int mw, int nw, int lane,
    v8f acc0[4], v8f acc1[4]) {
  v16bf a0 = load_frag_a(Asb + mw * 32, 32, 0, lane);
  v16bf a1 = load_frag_a(Asb + (mw + 16) * 32, 32, 0, lane);
#pragma unroll
  for (int t = 0; t < 4; ++t) {
    v16bf b = load_frag_b(Bsb + (nw + t * 16) * 32, 32, 0, lane);
    acc0[t] = WMMA_BF16(a0, b, acc0[t]);
    acc1[t] = WMMA_BF16(a1, b, acc1[t]);
  }
}

// ---------------------------------------------------------------------------
// Double-buffered block GEMM: block tile 128(M) x 128(N), K in 32-deep slices,
// unrolled two slices per iteration with explicit buffer identity so that the
// LDS fragment loads lower to ds_load_b128 (no generic/FLAT path).
// K must be a multiple of 64 (here K = 2048).
// ---------------------------------------------------------------------------
static __device__ __forceinline__ void gemm_pipeline(
    const __bf16* Ablk, int lda, const __bf16* Bblk, int ldb, int K,
    __bf16* As0, __bf16* As1, __bf16* Bs0, __bf16* Bs1,
    int tid, int lane, int mw, int nw, v8f acc0[4], v8f acc1[4]) {
  stage_tiles(Ablk, lda, Bblk, ldb, As0, Bs0, 0, tid);
  for (int kk = 0; kk < K; kk += 64) {
    // ---- phase 0: prefetch slice kk+32 -> buf1, compute slice kk from buf0
    stage_tiles(Ablk, lda, Bblk, ldb, As1, Bs1, kk + 32, tid);
    WAIT_ASYNC_4();            // slice kk landed; kk+32 still in flight
    __syncthreads();
    compute_slice(As0, Bs0, mw, nw, lane, acc0, acc1);
    __syncthreads();
    // ---- phase 1: prefetch slice kk+64 -> buf0, compute slice kk+32 from buf1
    if (kk + 64 < K) {
      stage_tiles(Ablk, lda, Bblk, ldb, As0, Bs0, kk + 64, tid);
      WAIT_ASYNC_4();
    } else {
      WAIT_ASYNC_0();
    }
    __syncthreads();
    compute_slice(As1, Bs1, mw, nw, lane, acc0, acc1);
    __syncthreads();
  }
}

// ---------------------------------------------------------------------------
// Conversion kernels
// ---------------------------------------------------------------------------
__global__ void k_cvt(const float* __restrict__ src, __bf16* __restrict__ dst, int n) {
  for (int i = blockIdx.x * blockDim.x + threadIdx.x; i < n; i += gridDim.x * blockDim.x)
    dst[i] = (__bf16)src[i];
}

// dst[N][K] = (bf16) src[K][N]^T   (write-coalesced)
__global__ void k_cvt_t(const float* __restrict__ src, __bf16* __restrict__ dst,
                        int K, int N) {
  int total = K * N;
  for (int i = blockIdx.x * blockDim.x + threadIdx.x; i < total; i += gridDim.x * blockDim.x) {
    int n = i / K, k = i - n * K;
    dst[i] = (__bf16)src[(size_t)k * N + n];
  }
}

// ---------------------------------------------------------------------------
// Fused QKV projection: [4096 x 2048] x [2048 x 3072], per-region weights,
// routed writes into Q [b,h,s,d], K [b,kv,s,d], V^T [b,kv,d,s].
// Grid: (4096/128, 3072/128) x 256 threads.
// ---------------------------------------------------------------------------
__global__ __launch_bounds__(256) void k_qkv(
    const __bf16* __restrict__ xb,
    const __bf16* __restrict__ WqT, const __bf16* __restrict__ WkT,
    const __bf16* __restrict__ WvT,
    const float* __restrict__ bq, const float* __restrict__ bk,
    const float* __restrict__ bv,
    __bf16* __restrict__ Qb, __bf16* __restrict__ Kb, __bf16* __restrict__ Vt) {
  __shared__ __align__(16) __bf16 As0[128 * 32];
  __shared__ __align__(16) __bf16 As1[128 * 32];
  __shared__ __align__(16) __bf16 Bs0[128 * 32];
  __shared__ __align__(16) __bf16 Bs1[128 * 32];

  int tid = threadIdx.x;
  int lane = tid & 31, w = tid >> 5;
  int m0 = blockIdx.x * 128;
  int n0 = blockIdx.y * 128;
  int mw = (w & 3) * 32, nw = (w >> 2) * 64;

  const __bf16* BT;
  const float* bias;
  int nloc;
  if (n0 < 2048)      { BT = WqT; bias = bq; nloc = n0; }
  else if (n0 < 2560) { BT = WkT; bias = bk; nloc = n0 - 2048; }
  else                { BT = WvT; bias = bv; nloc = n0 - 2560; }

  v8f acc0[4], acc1[4];
#pragma unroll
  for (int t = 0; t < 4; ++t) { acc0[t] = {}; acc1[t] = {}; }

  gemm_pipeline(xb + (size_t)m0 * 2048, 2048, BT + (size_t)nloc * 2048, 2048, 2048,
                As0, As1, Bs0, Bs1, tid, lane, mw, nw, acc0, acc1);

  int half = lane >> 4, nn = lane & 15;
  v8f* accs[2] = {acc0, acc1};
#pragma unroll
  for (int g = 0; g < 2; ++g) {
    int rowbase = m0 + mw + g * 16;
#pragma unroll
    for (int t = 0; t < 4; ++t) {
#pragma unroll
      for (int r = 0; r < 8; ++r) {
        int row = rowbase + r + 8 * half;   // C-frag: VGPR r -> M = r + 8*half
        int col = n0 + nw + t * 16 + nn;    // lane -> N
        float val = accs[g][t][r] + bias[nloc + nw + t * 16 + nn];
        __bf16 q = (__bf16)val;
        int bb = row >> 11, ss = row & 2047;
        if (col < 2048) {
          int hh = col >> 6, d = col & 63;
          Qb[(((size_t)bb * 32 + hh) * 2048 + ss) * 64 + d] = q;
        } else if (col < 2560) {
          int c = col - 2048, kvh = c >> 6, d = c & 63;
          Kb[(((size_t)bb * 8 + kvh) * 2048 + ss) * 64 + d] = q;
        } else {
          int c = col - 2560, kvh = c >> 6, d = c & 63;
          Vt[(((size_t)bb * 8 + kvh) * 64 + d) * 2048 + ss] = q;
        }
      }
    }
  }
}

// ---------------------------------------------------------------------------
// Flash attention: 1 wave / block, 16 query rows, streaming 32-key blocks with
// online softmax; P re-striped via LDS into A-fragment layout for P*V WMMAs.
// Grid: B*H*(S/16) = 8192 blocks x 32 threads.
// ---------------------------------------------------------------------------
__global__ __launch_bounds__(32) void k_attn(
    const __bf16* __restrict__ Qb, const __bf16* __restrict__ Kb,
    const __bf16* __restrict__ Vt, __bf16* __restrict__ attnb) {
  __shared__ __align__(16) __bf16 Plds[16 * 32];

  int lane = threadIdx.x;
  int blk = blockIdx.x;
  int qt = blk & 127;
  int bh = blk >> 7;
  int b = bh >> 5, h = bh & 31, kv = h >> 2;
  int q0 = qt * 16;
  int half = lane >> 4, nn = lane & 15;

  const __bf16* Qbase = Qb + ((size_t)(b * 32 + h) * 2048 + q0) * 64;
  const __bf16* Kbase = Kb + (size_t)(b * 8 + kv) * 2048 * 64;
  const __bf16* Vbase = Vt + (size_t)(b * 8 + kv) * 64 * 2048;

  v16bf qa0 = load_frag_a(Qbase, 64, 0, lane);
  v16bf qa1 = load_frag_a(Qbase, 64, 32, lane);

  v8f O[4];
#pragma unroll
  for (int t = 0; t < 4; ++t) O[t] = {};
  float m[8], l[8];
#pragma unroll
  for (int r = 0; r < 8; ++r) { m[r] = -1e30f; l[r] = 0.0f; }

  const float scale = 0.125f;  // 1/sqrt(64)

  for (int j = 0; j < 2048; j += 32) {
    v8f s0 = {}, s1 = {};
    s0 = WMMA_BF16(qa0, load_frag_b(Kbase + (size_t)j * 64, 64, 0, lane), s0);
    s0 = WMMA_BF16(qa1, load_frag_b(Kbase + (size_t)j * 64, 64, 32, lane), s0);
    s1 = WMMA_BF16(qa0, load_frag_b(Kbase + (size_t)(j + 16) * 64, 64, 0, lane), s1);
    s1 = WMMA_BF16(qa1, load_frag_b(Kbase + (size_t)(j + 16) * 64, 64, 32, lane), s1);

#pragma unroll
    for (int r = 0; r < 8; ++r) {
      float v0 = s0[r] * scale, v1 = s1[r] * scale;
      float bm = fmaxf(v0, v1);
#pragma unroll
      for (int off = 1; off < 16; off <<= 1) bm = fmaxf(bm, __shfl_xor(bm, off, 32));
      float mn = fmaxf(m[r], bm);
      float alpha = __expf(m[r] - mn);
      m[r] = mn;
      float p0 = __expf(v0 - mn);
      float p1 = __expf(v1 - mn);
      float rs = p0 + p1;
#pragma unroll
      for (int off = 1; off < 16; off <<= 1) rs += __shfl_xor(rs, off, 32);
      l[r] = l[r] * alpha + rs;
#pragma unroll
      for (int t = 0; t < 4; ++t) O[t][r] *= alpha;
      int row = r + 8 * half;
      Plds[row * 32 + nn]      = (__bf16)p0;
      Plds[row * 32 + 16 + nn] = (__bf16)p1;
    }
    __syncthreads();

    v16bf pa = load_frag_a(Plds, 32, 0, lane);
#pragma unroll
    for (int t = 0; t < 4; ++t) {
      v16bf vb = load_frag_b(Vbase + (size_t)(t * 16) * 2048, 2048, j, lane);
      O[t] = WMMA_BF16(pa, vb, O[t]);
    }
    __syncthreads();
  }

#pragma unroll
  for (int r = 0; r < 8; ++r) {
    float inv = 1.0f / l[r];
    int row = q0 + r + 8 * half;
    size_t ob = ((size_t)b * 2048 + row) * 2048 + (size_t)h * 64;
#pragma unroll
    for (int t = 0; t < 4; ++t)
      attnb[ob + t * 16 + nn] = (__bf16)(O[t][r] * inv);
  }
}

// ---------------------------------------------------------------------------
// Output projection: out[4096 x 2048] = attnb x Wo + bo, f32 output.
// Grid: (4096/128, 2048/128) x 256 threads.
// ---------------------------------------------------------------------------
__global__ __launch_bounds__(256) void k_oproj(
    const __bf16* __restrict__ attnb, const __bf16* __restrict__ WoT,
    const float* __restrict__ bo, float* __restrict__ out) {
  __shared__ __align__(16) __bf16 As0[128 * 32];
  __shared__ __align__(16) __bf16 As1[128 * 32];
  __shared__ __align__(16) __bf16 Bs0[128 * 32];
  __shared__ __align__(16) __bf16 Bs1[128 * 32];

  int tid = threadIdx.x;
  int lane = tid & 31, w = tid >> 5;
  int m0 = blockIdx.x * 128;
  int n0 = blockIdx.y * 128;
  int mw = (w & 3) * 32, nw = (w >> 2) * 64;

  v8f acc0[4], acc1[4];
#pragma unroll
  for (int t = 0; t < 4; ++t) { acc0[t] = {}; acc1[t] = {}; }

  gemm_pipeline(attnb + (size_t)m0 * 2048, 2048, WoT + (size_t)n0 * 2048, 2048, 2048,
                As0, As1, Bs0, Bs1, tid, lane, mw, nw, acc0, acc1);

  int half = lane >> 4, nn = lane & 15;
  v8f* accs[2] = {acc0, acc1};
#pragma unroll
  for (int g = 0; g < 2; ++g) {
    int rowbase = m0 + mw + g * 16;
#pragma unroll
    for (int t = 0; t < 4; ++t) {
#pragma unroll
      for (int r = 0; r < 8; ++r) {
        int row = rowbase + r + 8 * half;
        int col = n0 + nw + t * 16 + nn;
        out[(size_t)row * 2048 + col] = accs[g][t][r] + bo[col];
      }
    }
  }
}

// ---------------------------------------------------------------------------
extern "C" void kernel_launch(void* const* d_in, const int* in_sizes, int n_in,
                              void* d_out, int out_size, void* d_ws, size_t ws_size,
                              hipStream_t stream) {
  const float* x  = (const float*)d_in[0];
  const float* Wq = (const float*)d_in[1];
  const float* bq = (const float*)d_in[2];
  const float* Wk = (const float*)d_in[3];
  const float* bk = (const float*)d_in[4];
  const float* Wv = (const float*)d_in[5];
  const float* bv = (const float*)d_in[6];
  const float* Wo = (const float*)d_in[7];
  const float* bo = (const float*)d_in[8];
  float* out = (float*)d_out;

  char* ws = (char*)d_ws;
  size_t off = 0;
  __bf16* xb    = (__bf16*)(ws + off); off += (size_t)8388608 * 2;  // 4096x2048
  __bf16* WqT   = (__bf16*)(ws + off); off += (size_t)4194304 * 2;  // 2048x2048
  __bf16* WkT   = (__bf16*)(ws + off); off += (size_t)1048576 * 2;  // 512x2048
  __bf16* WvT   = (__bf16*)(ws + off); off += (size_t)1048576 * 2;  // 512x2048
  __bf16* WoT   = (__bf16*)(ws + off); off += (size_t)4194304 * 2;  // 2048x2048
  __bf16* Qb    = (__bf16*)(ws + off); off += (size_t)8388608 * 2;  // [b,h,s,d]
  __bf16* Kb    = (__bf16*)(ws + off); off += (size_t)2097152 * 2;  // [b,kv,s,d]
  __bf16* Vt    = (__bf16*)(ws + off); off += (size_t)2097152 * 2;  // [b,kv,d,s]
  __bf16* attnb = (__bf16*)(ws + off); off += (size_t)8388608 * 2;  // [b,s,h*d]

  // 1) conversions / transposes to bf16
  k_cvt<<<4096, 256, 0, stream>>>(x, xb, 8388608);
  k_cvt_t<<<4096, 256, 0, stream>>>(Wq, WqT, 2048, 2048);
  k_cvt_t<<<1024, 256, 0, stream>>>(Wk, WkT, 2048, 512);
  k_cvt_t<<<1024, 256, 0, stream>>>(Wv, WvT, 2048, 512);
  k_cvt_t<<<4096, 256, 0, stream>>>(Wo, WoT, 2048, 2048);

  // 2) fused QKV projection (WMMA, async-LDS pipelined)
  k_qkv<<<dim3(32, 24), 256, 0, stream>>>(xb, WqT, WkT, WvT, bq, bk, bv, Qb, Kb, Vt);

  // 3) flash attention (WMMA): B*H*(S/16) blocks, 1 wave each
  k_attn<<<8192, 32, 0, stream>>>(Qb, Kb, Vt, attnb);

  // 4) output projection (WMMA, async-LDS pipelined) + bias, f32 out
  k_oproj<<<dim3(32, 16), 256, 0, stream>>>(attnb, WoT, bo, out);
}
